// SAVPCell_50440095924407
// MI455X (gfx1250) — matrix operand order repaired and checked by reading, not compiled, not run on hardware
//
#include <hip/hip_runtime.h>

// SAVP "CDNA kernels": per-batch 5x5 cross-correlation via one
// v_wmma_f32_16x16x32_f16 per 16-wide output tile (25 taps zero-padded to K=32).
// images  [B=64, H=128, W=128, C=3]  f32
// kernels [B=64, 5, 5, N=10]         f32
// out     [N=10, B=64, C=3, H=128, W=128] f32
//
// Block = (b, h0..h0+7), 8 waves. Padded f16 image slab staged in LDS
// (12 rows x 132 cols x 4 halves/pixel), zero padding materialized, so the
// im2col A-operand build is branchless ds_load_b64 + v_perm_b32 only.

typedef __attribute__((ext_vector_type(16))) _Float16 v16h;
typedef __attribute__((ext_vector_type(8)))  float    v8f;
typedef __attribute__((ext_vector_type(8)))  unsigned v8u;

#define B_ 64
#define H_ 128
#define W_ 128
#define C_ 3
#define N_ 10
#define SN (B_ * C_ * H_ * W_)   // n-stride of output, floats

#define ROWS 12                  // rows h0-2 .. h0+9
#define PW   132                 // padded width: w = -2 .. 129
#define NPIX (ROWS * PW)         // 1584 pixels
#define ZSLOT NPIX               // index of the always-zero pixel (taps k>=25)

static __device__ __forceinline__ unsigned pk16(float a, float b) {
  union { _Float16 h[2]; unsigned u; } t;
  t.h[0] = (_Float16)a;
  t.h[1] = (_Float16)b;
  return t.u;
}

__global__ __launch_bounds__(256) void savp_cdna_wmma(
    const float* __restrict__ images,
    const float* __restrict__ kern,
    float* __restrict__ out)
{
  __shared__ uint2 sTile[NPIX + 1];   // 12680 B: pixel = {c0|c1, c2|0} f16

  const int tid = threadIdx.x;
  const int b   = blockIdx.x >> 4;          // 64 batches
  const int h0  = (blockIdx.x & 15) << 3;   // 16 row-groups of 8

  const float* imgB = images + (size_t)b * (H_ * W_ * C_);

  // ---- Cooperative fill: f32 -> f16 once, padding written as zeros ----
  for (int i = tid; i < NPIX; i += 256) {
    const unsigned row = (unsigned)i / (unsigned)PW;
    const int col = i - (int)row * PW;
    const int hh  = h0 + (int)row - 2;
    const int ww  = col - 2;
    float x = 0.f, y = 0.f, z = 0.f;
    if ((unsigned)hh < (unsigned)H_ && (unsigned)ww < (unsigned)W_) {
      const float* p = imgB + (hh * W_ + ww) * C_;
      x = p[0]; y = p[1]; z = p[2];
    }
    sTile[i] = make_uint2(pk16(x, y), pk16(z, 0.f));
  }
  if (tid == 0) sTile[ZSLOT] = make_uint2(0u, 0u);
  __syncthreads();

  // ---- Per-wave compute: wave = output row h = h0 + wid ----
  const int lane = tid & 31;
  const int wid  = tid >> 5;
  const int h    = h0 + wid;
  const int m    = lane & 15;   // A row (w offset) / B,D column (n)
  const int hi   = lane >> 4;

  // B operand: kernels, 32(K) x 16(n), f16. Branchless (clamped) loads.
  const float* kb = kern + (size_t)b * (25 * N_);
  v16h bw = {};
  #pragma unroll
  for (int e = 0; e < 16; ++e) {
    const int  k  = hi * 16 + e;                // lane col = m, half e -> K
    const bool ok = (k < 25) & (m < N_);
    const float f = kb[ok ? k * N_ + m : 0];
    bw[e] = ok ? (_Float16)f : (_Float16)0.f;
  }

  // Per-tap LDS byte offsets for the A operand (im2col), hoisted.
  // A layout: lane row = m; half e -> K = 16*hi+e.
  int ldsOff[16];
  #pragma unroll
  for (int e = 0; e < 16; ++e) {
    const int k  = hi * 16 + e;
    const int kh = (k * 205) >> 10;             // k/5 for k<32
    const int kw = k - kh * 5;
    ldsOff[e] = (k < 25) ? ((wid + kh) * PW + m + kw) * 8 : ZSLOT * 8;
  }

  // Output base for this lane: n = m (masked to n<10), w = wt*16 + hi*8 + r.
  float* q0 = out + (size_t)m * SN
            + ((size_t)(b * C_) * H_ + h) * W_ + hi * 8;
  const char* sBase = (const char*)&sTile[0];

  #pragma unroll
  for (int wt = 0; wt < 8; ++wt) {
    // 16 unconditional ds_load_b64; wt advances via DS immediate offset.
    uint2 d[16];
    #pragma unroll
    for (int e = 0; e < 16; ++e)
      d[e] = *(const uint2*)(sBase + ldsOff[e] + wt * 128);

    // Repack: one v_perm_b32 per A VGPR per channel.
    v8u a0, a1, a2;
    #pragma unroll
    for (int v = 0; v < 8; ++v) {
      a0[v] = __builtin_amdgcn_perm(d[2*v+1].x, d[2*v].x, 0x05040100u); // c0
      a1[v] = __builtin_amdgcn_perm(d[2*v+1].x, d[2*v].x, 0x07060302u); // c1
      a2[v] = __builtin_amdgcn_perm(d[2*v+1].y, d[2*v].y, 0x05040100u); // c2
    }

    v8f c0 = {}, c1 = {}, c2 = {};
    c0 = __builtin_amdgcn_wmma_f32_16x16x32_f16(
        false, __builtin_bit_cast(v16h, a0), false, bw, (short)0, c0, false, false);
    c1 = __builtin_amdgcn_wmma_f32_16x16x32_f16(
        false, __builtin_bit_cast(v16h, a1), false, bw, (short)0, c1, false, false);
    c2 = __builtin_amdgcn_wmma_f32_16x16x32_f16(
        false, __builtin_bit_cast(v16h, a2), false, bw, (short)0, c2, false, false);

    if (m < N_) {  // lanes n>=10 masked; one exec toggle per tile
      *(v8f*)(q0 + wt * 16)               = c0;
      *(v8f*)(q0 + H_ * W_ + wt * 16)     = c1;
      *(v8f*)(q0 + 2 * H_ * W_ + wt * 16) = c2;
    }
  }
}

extern "C" void kernel_launch(void* const* d_in, const int* in_sizes, int n_in,
                              void* d_out, int out_size, void* d_ws, size_t ws_size,
                              hipStream_t stream) {
  (void)in_sizes; (void)n_in; (void)d_ws; (void)ws_size; (void)out_size;
  const float* images = (const float*)d_in[0];
  const float* kern   = (const float*)d_in[1];
  float*       out    = (float*)d_out;
  dim3 grid(B_ * (H_ / 8));   // 1024 blocks: (b, 8-row group)
  dim3 block(256);            // 8 waves
  savp_cdna_wmma<<<grid, block, 0, stream>>>(images, kern, out);
}
